// _ColGroupedLinear_74612171866404
// MI455X (gfx1250) — matrix operand order repaired and checked by reading, not compiled, not run on hardware
//
#include <hip/hip_runtime.h>

// CDNA5 / gfx1250, wave32. bf16x3 split-precision VQ-dequant GEMM.
typedef __attribute__((ext_vector_type(16))) __bf16 v16bf;
typedef __attribute__((ext_vector_type(8)))  __bf16 v8bf;
typedef __attribute__((ext_vector_type(8)))  float  v8f;

#define KC   32      // K-chunk per stage (one bf16 WMMA depth)
#define KCP  40      // padded K stride in LDS (bf16 elems) = 80B rows, 16B aligned
#define MT   128     // workgroup M tile
#define NT   64      // workgroup N tile
#define GS   8       // VQ group size (centroid vector length)

// split f32 into bf16 hi + bf16 lo (x ~= hi + lo, lo holds the rounding residual)
#define SPL8(vh, vl, i, f)                         \
  do {                                             \
    float _f = (f);                                \
    __bf16 _h = (__bf16)_f;                        \
    (vh)[i] = _h;                                  \
    (vl)[i] = (__bf16)(_f - (float)_h);            \
  } while (0)

__device__ __forceinline__ v16bf load_afrag(const __bf16* tile, int row, int hs) {
  // A (16x32 bf16, MxK): lane<16 holds K 0..7 & 16..23 of row M=lane;
  // lane>=16 holds K 8..15 & 24..31.  tile is row-major [M][KCP].
  const __bf16* p = tile + row * KCP + hs * 8;
  v8bf a0 = *(const v8bf*)(p);
  v8bf a1 = *(const v8bf*)(p + 16);
  v16bf r;
#pragma unroll
  for (int i = 0; i < 8; ++i) { r[i] = a0[i]; r[8 + i] = a1[i]; }
  return r;
}

__device__ __forceinline__ v16bf load_bfrag(const __bf16* tile, int row, int hs) {
  // B (32x16 bf16, KxN): lane<16 holds K 0..7 & 16..23 of col N=lane; lane>=16 K 8..15 & 24..31.
  // tile is n-major [N][KCP] so this is two 16B reads per lane.
  const __bf16* p = tile + row * KCP + hs * 8;
  v8bf b0 = *(const v8bf*)(p);
  v8bf b1 = *(const v8bf*)(p + 16);
  v16bf r;
#pragma unroll
  for (int i = 0; i < 8; ++i) { r[i] = b0[i]; r[8 + i] = b1[i]; }
  return r;
}

__global__ __launch_bounds__(256) void vq_linear_bf16x3_kernel(
    const float* __restrict__ x,          // [M, K]
    const float* __restrict__ centroids,  // [n_centroids, GS]
    const int*   __restrict__ labels,     // [K, n_groups]
    const float* __restrict__ scale,      // [K]
    const float* __restrict__ bias,       // [N]
    float* __restrict__ out,              // [M, N]
    int M, int K, int N, int n_groups)
{
  __shared__ __bf16 sXhi[MT * KCP];
  __shared__ __bf16 sXlo[MT * KCP];
  __shared__ __bf16 sWhi[NT * KCP];
  __shared__ __bf16 sWlo[NT * KCP];

  const int tid  = threadIdx.x;
  const int lane = tid & 31;
  const int l16  = lane & 15;
  const int hs   = lane >> 4;     // lane half-select for WMMA fragments
  const int wv   = tid >> 5;      // wave 0..7
  const int wm   = wv >> 1;       // 0..3  -> 32-row strip
  const int wn   = wv & 1;        // 0..1  -> 32-col strip

  const int m0 = blockIdx.y * MT;
  const int n0 = blockIdx.x * NT;
  const int g0 = n0 / GS;         // first label-group of this N tile

  // staging roles
  const int xrow = tid >> 1;      // 0..127 : x row within tile
  const int xh   = tid & 1;       // which 16-wide K half
  const int wk   = tid >> 3;      // 0..31  : k row for W dequant
  const int wg   = tid & 7;       // 0..7   : group (8 cols) within N tile

  v8f acc[2][2];
#pragma unroll
  for (int i = 0; i < 2; ++i)
#pragma unroll
    for (int j = 0; j < 2; ++j)
#pragma unroll
      for (int r = 0; r < 8; ++r) acc[i][j][r] = 0.0f;

  float bv[2];
#pragma unroll
  for (int ns = 0; ns < 2; ++ns)
    bv[ns] = bias[n0 + wn * 32 + ns * 16 + l16];

  const int nk = K / KC;
  for (int kc = 0; kc < nk; ++kc) {
    // ---- stage X: 128x32 f32 -> bf16 hi/lo tiles in LDS ----
    {
      const float* xp = x + (size_t)(m0 + xrow) * K + kc * KC + xh * 16;
      const float4* xv = (const float4*)xp;
      float4 v0 = xv[0], v1 = xv[1], v2 = xv[2], v3 = xv[3];
      if (kc + 1 < nk) __builtin_prefetch(xp + KC, 0, 1);  // global_prefetch_b8
      v8bf h0, h1, l0, l1;
      SPL8(h0, l0, 0, v0.x); SPL8(h0, l0, 1, v0.y); SPL8(h0, l0, 2, v0.z); SPL8(h0, l0, 3, v0.w);
      SPL8(h0, l0, 4, v1.x); SPL8(h0, l0, 5, v1.y); SPL8(h0, l0, 6, v1.z); SPL8(h0, l0, 7, v1.w);
      SPL8(h1, l1, 0, v2.x); SPL8(h1, l1, 1, v2.y); SPL8(h1, l1, 2, v2.z); SPL8(h1, l1, 3, v2.w);
      SPL8(h1, l1, 4, v3.x); SPL8(h1, l1, 5, v3.y); SPL8(h1, l1, 6, v3.z); SPL8(h1, l1, 7, v3.w);
      __bf16* dh = &sXhi[xrow * KCP + xh * 16];
      __bf16* dl = &sXlo[xrow * KCP + xh * 16];
      *(v8bf*)(dh)     = h0;
      *(v8bf*)(dh + 8) = h1;
      *(v8bf*)(dl)     = l0;
      *(v8bf*)(dl + 8) = l1;
    }

    // ---- stage W: gather-dequant 32x64 from codebook -> bf16 hi/lo (n-major) ----
    {
      const int kg = kc * KC + wk;
      const int lbl = labels[(size_t)kg * n_groups + g0 + wg];
      const float* cb = centroids + (size_t)(unsigned)lbl * GS;
      float4 c0 = *(const float4*)(cb);
      float4 c1 = *(const float4*)(cb + 4);
      const float invs = 1.0f / fmaxf(scale[kg], 1e-8f);
#define WST(j, comp)                                          \
      do {                                                    \
        float _w  = (comp) * invs;                            \
        __bf16 _h = (__bf16)_w;                               \
        sWhi[(wg * 8 + (j)) * KCP + wk] = _h;                 \
        sWlo[(wg * 8 + (j)) * KCP + wk] = (__bf16)(_w - (float)_h); \
      } while (0)
      WST(0, c0.x); WST(1, c0.y); WST(2, c0.z); WST(3, c0.w);
      WST(4, c1.x); WST(5, c1.y); WST(6, c1.z); WST(7, c1.w);
#undef WST
    }

    __syncthreads();

    // ---- compute: each wave 2x2 WMMA tiles, 3 products (hi*hi + hi*lo + lo*hi) ----
    v16bf bh[2], bl[2];
#pragma unroll
    for (int ns = 0; ns < 2; ++ns) {
      bh[ns] = load_bfrag(sWhi, wn * 32 + ns * 16 + l16, hs);
      bl[ns] = load_bfrag(sWlo, wn * 32 + ns * 16 + l16, hs);
    }
#pragma unroll
    for (int ms = 0; ms < 2; ++ms) {
      v16bf ah = load_afrag(sXhi, wm * 32 + ms * 16 + l16, hs);
      v16bf al = load_afrag(sXlo, wm * 32 + ms * 16 + l16, hs);
#pragma unroll
      for (int ns = 0; ns < 2; ++ns) {
        acc[ms][ns] = __builtin_amdgcn_wmma_f32_16x16x32_bf16(
            false, ah, false, bh[ns], (short)0, acc[ms][ns], false, false);
        acc[ms][ns] = __builtin_amdgcn_wmma_f32_16x16x32_bf16(
            false, ah, false, bl[ns], (short)0, acc[ms][ns], false, false);
        acc[ms][ns] = __builtin_amdgcn_wmma_f32_16x16x32_bf16(
            false, al, false, bh[ns], (short)0, acc[ms][ns], false, false);
      }
    }

    __syncthreads();
  }

  // ---- epilogue: C layout is VGPR r -> row (hs*8 + r), col l16 ----
#pragma unroll
  for (int ms = 0; ms < 2; ++ms) {
#pragma unroll
    for (int ns = 0; ns < 2; ++ns) {
      const int col  = n0 + wn * 32 + ns * 16 + l16;
      const int rowb = m0 + wm * 32 + ms * 16 + hs * 8;
#pragma unroll
      for (int r = 0; r < 8; ++r) {
        out[(size_t)(rowb + r) * N + col] = acc[ms][ns][r] + bv[ns];
      }
    }
  }
}

extern "C" void kernel_launch(void* const* d_in, const int* in_sizes, int n_in,
                              void* d_out, int out_size, void* d_ws, size_t ws_size,
                              hipStream_t stream) {
  const float* x         = (const float*)d_in[0];
  const float* centroids = (const float*)d_in[1];
  const int*   labels    = (const int*)d_in[2];
  const float* scale     = (const float*)d_in[3];
  const float* bias      = (const float*)d_in[4];
  float*       out       = (float*)d_out;

  const int K        = in_sizes[3];             // in_f (4096)
  const int N        = in_sizes[4];             // out_f (4096)
  const int n_groups = in_sizes[2] / K;         // 512
  const int M        = in_sizes[0] / K;         // batch*seq (8192)

  dim3 grid(N / NT, M / MT);                    // (64, 64)
  vq_linear_bf16x3_kernel<<<grid, 256, 0, stream>>>(
      x, centroids, labels, scale, bias, out, M, K, N, n_groups);
}
